// FiBiNET_15144054686441
// MI455X (gfx1250) — matrix operand order repaired and checked by reading, not compiled, not run on hardware
//
#include <hip/hip_runtime.h>
#include <hip/hip_bf16.h>

typedef __attribute__((ext_vector_type(16))) __bf16 v16bf;
typedef __attribute__((ext_vector_type(8)))  __bf16 v8bf;
typedef __attribute__((ext_vector_type(8)))  float  v8f;

#define B_   2048
#define F_   26
#define E_   64
#define D_   13
#define P_   325
#define PP_  650
#define KC_  41600   // 2*P*E  (c columns, K of layer0 handled by WMMA)
#define N0_  1024
#define N1_  512
#define N2_  256

static __device__ inline v16bf join8(v8bf lo, v8bf hi) {
  v16bf r;
#pragma unroll
  for (int i = 0; i < 8; ++i) { r[i] = lo[i]; r[8 + i] = hi[i]; }
  return r;
}
static __device__ inline v8f zero8() {
  v8f z;
#pragma unroll
  for (int i = 0; i < 8; ++i) z[i] = 0.0f;
  return z;
}
// pair index -> (i,j) for combinations(range(26), 2), lexicographic
static __device__ inline void pair_ij(int p, int& i, int& j) {
  int ii = 0, rem = p, span = F_ - 1;
  while (rem >= span) { rem -= span; ++ii; --span; }
  i = ii; j = ii + 1 + rem;
}

// CDNA5 async DMA: global -> LDS, 16 bytes per lane, tracked by ASYNCcnt.
static __device__ inline void async_copy_b128(const void* gptr, void* lptr) {
  uint32_t lds_off = (uint32_t)(size_t)lptr;          // addr[31:0] == LDS offset
  uint64_t gaddr   = (uint64_t)(size_t)gptr;
  asm volatile("global_load_async_to_lds_b128 %0, %1, off"
               :: "v"(lds_off), "v"(gaddr) : "memory");
}
static __device__ inline void wait_async0() {
  asm volatile("s_wait_asynccnt 0x0" ::: "memory");
}

// ---------------------------------------------------------------------------
// fp32 [.., K, N] -> bf16 [.., N, K] (transpose so B-fragments are contiguous)
// ---------------------------------------------------------------------------
__global__ void __launch_bounds__(256)
k_convert_transpose(const float* __restrict__ W, __bf16* __restrict__ Wt,
                    int K, int N, long long total) {
  long long stride = (long long)gridDim.x * blockDim.x;
  for (long long idx = blockIdx.x * (long long)blockDim.x + threadIdx.x;
       idx < total; idx += stride) {
    long long mat = idx / ((long long)K * N);
    long long rem = idx - mat * (long long)K * N;
    int k = (int)(rem / N);
    int n = (int)(rem - (long long)k * N);
    Wt[mat * (long long)K * N + (long long)n * K + k] = (__bf16)W[idx];
  }
}

// ---------------------------------------------------------------------------
// Per-sample: embedding gather, SENET excitation, LR logit, bf16 emb/Vemb
// ---------------------------------------------------------------------------
__global__ void __launch_bounds__(128)
k_prep(const int* __restrict__ ids, const float* __restrict__ dense,
       const float* __restrict__ emb_table, const float* __restrict__ lr_table,
       const float* __restrict__ lr_dense_w, const float* __restrict__ lr_bias,
       const float* __restrict__ w1, const float* __restrict__ w2,
       __bf16* __restrict__ emb_out, __bf16* __restrict__ vemb_out,
       float* __restrict__ lr_logit) {
  __shared__ float s_emb[F_ * E_];
  __shared__ int   s_id[F_];
  __shared__ float s_Z[F_], s_t[8], s_A[F_], s_lr[40];
  const int b = blockIdx.x, tid = threadIdx.x;
  if (tid < F_) s_id[tid] = ids[b * F_ + tid];
  __syncthreads();
  for (int idx = tid; idx < F_ * E_; idx += 128) {
    int f = idx >> 6, e = idx & 63;
    s_emb[idx] = emb_table[(size_t)s_id[f] * E_ + e];
  }
  __syncthreads();
  if (tid < F_) {                      // squeeze (mean over E) + LR gather
    float s = 0.f;
    for (int e = 0; e < E_; ++e) s += s_emb[tid * E_ + e];
    s_Z[tid]  = s * (1.0f / E_);
    s_lr[tid] = lr_table[s_id[tid]];
  } else if (tid < F_ + D_) {
    int d = tid - F_;
    s_lr[tid] = dense[b * D_ + d] * lr_dense_w[d];
  } else if (tid < 40) {
    s_lr[tid] = 0.f;
  }
  __syncthreads();
  if (tid < 8) {                       // excitation layer 1 (26 -> 8), ReLU
    float s = 0.f;
    for (int f = 0; f < F_; ++f) s += s_Z[f] * w1[f * 8 + tid];
    s_t[tid] = fmaxf(s, 0.f);
  }
  __syncthreads();
  if (tid < F_) {                      // excitation layer 2 (8 -> 26), ReLU
    float s = 0.f;
    for (int r = 0; r < 8; ++r) s += s_t[r] * w2[r * F_ + tid];
    s_A[tid] = fmaxf(s, 0.f);
  }
  if (tid == 0) {
    float s = lr_bias[0];
    for (int t = 0; t < F_ + D_; ++t) s += s_lr[t];
    lr_logit[b] = s;
  }
  __syncthreads();
  for (int idx = tid; idx < F_ * E_; idx += 128) {
    int f = idx >> 6;
    float v = s_emb[idx];
    emb_out [(size_t)b * (F_ * E_) + idx] = (__bf16)v;
    vemb_out[(size_t)b * (F_ * E_) + idx] = (__bf16)(v * s_A[f]);
  }
}

// ---------------------------------------------------------------------------
// Bilinear field interaction: c[b, pp*64+n] = ((x_i @ W_p) * x_j)[b,n], bf16
// One wave per (16-row tile, pair-slot). 8 x v_wmma_f32_16x16x32_bf16 / wave.
// ---------------------------------------------------------------------------
__global__ void __launch_bounds__(128)
k_bilinear(const __bf16* __restrict__ emb, const __bf16* __restrict__ vemb,
           const __bf16* __restrict__ Wbt,   // [P][N=64][K=64] bf16
           __bf16* __restrict__ cbuf) {
  const int pp = blockIdx.y * blockDim.y + threadIdx.y;
  if (pp >= PP_) return;
  const int pair = (pp < P_) ? pp : pp - P_;
  const __bf16* src = (pp < P_) ? emb : vemb;
  int fi, fj; pair_ij(pair, fi, fj);
  const int rt   = blockIdx.x;          // 16-row batch tile
  const int lane = threadIdx.x;         // 0..31 (wave32)
  const int colL = lane & 15;
  const int hi   = lane >> 4;

  // A fragments (rows = batch rows, K = embedding dim 64 -> two K=32 chunks)
  v16bf afr[2];
  {
    const int brow = rt * 16 + colL;
    const __bf16* abase = src + ((size_t)brow * F_ + fi) * E_;
#pragma unroll
    for (int kc = 0; kc < 2; ++kc) {
      v8bf p0 = *(const v8bf*)(abase + kc * 32 + hi * 8);
      v8bf p1 = *(const v8bf*)(abase + kc * 32 + 16 + hi * 8);
      afr[kc] = join8(p0, p1);
    }
  }
  const __bf16* wbase = Wbt + (size_t)pair * E_ * E_;
#pragma unroll
  for (int nt = 0; nt < 4; ++nt) {
    v8f acc = zero8();
#pragma unroll
    for (int kc = 0; kc < 2; ++kc) {
      const __bf16* bb = wbase + (size_t)(nt * 16 + colL) * E_ + kc * 32 + hi * 8;
      v8bf p0 = *(const v8bf*)(bb);
      v8bf p1 = *(const v8bf*)(bb + 16);
      v16bf bfr = join8(p0, p1);
      acc = __builtin_amdgcn_wmma_f32_16x16x32_bf16(
          false, afr[kc], false, bfr, (short)0, acc, false, false);
    }
    const int n = nt * 16 + colL;
    // batch the 8 v_j loads so they pipeline in one clause (no per-element wait)
    __bf16 vj[8];
#pragma unroll
    for (int r = 0; r < 8; ++r) {
      const int b = rt * 16 + hi * 8 + r;
      vj[r] = src[((size_t)b * F_ + fj) * E_ + n];
    }
#pragma unroll
    for (int r = 0; r < 8; ++r) {
      const int b = rt * 16 + hi * 8 + r;
      cbuf[(size_t)b * KC_ + pp * 64 + n] = (__bf16)(acc[r] * (float)vj[r]);
    }
  }
}

// ---------------------------------------------------------------------------
// GEMM + bias (+ dense-tail for layer 0) + ReLU, bf16 in / bf16 out.
// Block 256 = 8 waves; tile 128(M) x 64(N); wave computes 32x32 (4 frags).
// Double-buffered LDS staged with global_load_async_to_lds_b128 (ASYNCcnt):
// chunk i+1 DMA overlaps chunk i WMMAs; one barrier per K-chunk.
// ---------------------------------------------------------------------------
__global__ void __launch_bounds__(256)
k_gemm(const __bf16* __restrict__ A, const __bf16* __restrict__ Bt,
       const float* __restrict__ bias, __bf16* __restrict__ out,
       int M, int N, int K,
       const float* __restrict__ dense, const float* __restrict__ Wtail,
       int addDense) {
  __shared__ __bf16 sA[2][128 * 40];
  __shared__ __bf16 sB[2][64 * 40];
  const int tid    = threadIdx.x;
  const int blockM = blockIdx.y * 128;
  const int blockN = blockIdx.x * 64;
  const int lane   = tid & 31;
  const int wave   = tid >> 5;
  const int wm     = wave >> 1;       // 0..3
  const int wn     = wave & 1;        // 0..1
  const int colL   = lane & 15;
  const int hi     = lane >> 4;

  v8f acc[4];
#pragma unroll
  for (int i = 0; i < 4; ++i) acc[i] = zero8();

  const int ar  = tid >> 1;           // A staging row 0..127
  const int aks = (tid & 1) * 16;     // k sub-offset 0/16
  const int bn  = tid >> 2;           // B staging row (output col) 0..63
  const int bks = (tid & 3) * 8;      // k sub-offset 0/8/16/24

  const __bf16* gA = A  + (size_t)(blockM + ar) * K + aks;
  const __bf16* gB = Bt + (size_t)(blockN + bn) * K + bks;

  const int nchunks = K / 32;
  // prologue: DMA chunk 0 into buffer 0
  async_copy_b128(gA,     &sA[0][ar * 40 + aks]);
  async_copy_b128(gA + 8, &sA[0][ar * 40 + aks + 8]);
  async_copy_b128(gB,     &sB[0][bn * 40 + bks]);

  for (int c = 0; c < nchunks; ++c) {
    const int buf = c & 1;
    wait_async0();                    // my chunk-c DMAs landed in LDS
    __syncthreads();                  // everyone's landed; prior reads of other buf done
    if (c + 1 < nchunks) {            // DMA chunk c+1 into the other buffer
      const int nb = buf ^ 1, kk = (c + 1) * 32;
      async_copy_b128(gA + kk,     &sA[nb][ar * 40 + aks]);
      async_copy_b128(gA + kk + 8, &sA[nb][ar * 40 + aks + 8]);
      async_copy_b128(gB + kk,     &sB[nb][bn * 40 + bks]);
    }
    v16bf afr[2], bfr[2];
#pragma unroll
    for (int mt = 0; mt < 2; ++mt) {
      const int rbase = (wm * 32 + mt * 16 + colL) * 40;
      v8bf p0 = *(const v8bf*)(&sA[buf][rbase + hi * 8]);
      v8bf p1 = *(const v8bf*)(&sA[buf][rbase + 16 + hi * 8]);
      afr[mt] = join8(p0, p1);
    }
#pragma unroll
    for (int nt = 0; nt < 2; ++nt) {
      const int nb2 = (wn * 32 + nt * 16 + colL) * 40 + hi * 8;
      v8bf p0 = *(const v8bf*)(&sB[buf][nb2]);
      v8bf p1 = *(const v8bf*)(&sB[buf][nb2 + 16]);
      bfr[nt] = join8(p0, p1);
    }
#pragma unroll
    for (int mt = 0; mt < 2; ++mt)
#pragma unroll
      for (int nt = 0; nt < 2; ++nt)
        acc[mt * 2 + nt] = __builtin_amdgcn_wmma_f32_16x16x32_bf16(
            false, afr[mt], false, bfr[nt], (short)0, acc[mt * 2 + nt],
            false, false);
  }

#pragma unroll
  for (int mt = 0; mt < 2; ++mt) {
#pragma unroll
    for (int nt = 0; nt < 2; ++nt) {
      const int col = blockN + wn * 32 + nt * 16 + colL;
      const float bcol = bias[col];
#pragma unroll
      for (int r = 0; r < 8; ++r) {
        const int row = blockM + wm * 32 + mt * 16 + hi * 8 + r;
        float v = acc[mt * 2 + nt][r] + bcol;
        if (addDense) {
          const float* dr = dense + row * D_;
#pragma unroll
          for (int d = 0; d < D_; ++d) v += dr[d] * Wtail[d * N + col];
        }
        v = fmaxf(v, 0.0f);
        out[(size_t)row * N + col] = (__bf16)v;
      }
    }
  }
}

// ---------------------------------------------------------------------------
// Final layer (256 -> 1) + LR logit + sigmoid + BCE mean
// ---------------------------------------------------------------------------
__global__ void k_zero_loss(float* out) { if (threadIdx.x == 0) out[B_] = 0.f; }

__global__ void __launch_bounds__(256)
k_head(const __bf16* __restrict__ h3, const float* __restrict__ W3,
       const float* __restrict__ b3, const float* __restrict__ lr_logit,
       const int* __restrict__ label, float* __restrict__ out) {
  __shared__ float s_part[8];
  const int tid = threadIdx.x, lane = tid & 31, wave = tid >> 5;
  const int row = blockIdx.x * 8 + wave;
  float dot = 0.f;
  for (int k = lane; k < N2_; k += 32)
    dot += (float)h3[(size_t)row * N2_ + k] * W3[k];
  for (int off = 16; off > 0; off >>= 1) dot += __shfl_down(dot, off, 32);
  if (lane == 0) {
    float logit = dot + b3[0] + lr_logit[row];
    float y = 1.0f / (1.0f + expf(-logit));
    out[row] = y;
    float yc = fminf(fmaxf(y, 1e-7f), 1.0f - 1e-7f);
    float yt = (float)label[row];
    s_part[wave] = -(yt * logf(yc) + (1.0f - yt) * logf(1.0f - yc));
  }
  __syncthreads();
  if (tid == 0) {
    float s = 0.f;
    for (int w = 0; w < 8; ++w) s += s_part[w];
    atomicAdd(out + B_, s * (1.0f / B_));
  }
}

// ---------------------------------------------------------------------------
extern "C" void kernel_launch(void* const* d_in, const int* in_sizes, int n_in,
                              void* d_out, int out_size, void* d_ws, size_t ws_size,
                              hipStream_t stream) {
  (void)in_sizes; (void)n_in; (void)out_size;
  const int*   sparse_ids = (const int*)  d_in[0];
  const float* dense      = (const float*)d_in[1];
  const int*   label      = (const int*)  d_in[2];
  const float* emb_table  = (const float*)d_in[3];
  const float* lr_table   = (const float*)d_in[4];
  const float* lr_dense_w = (const float*)d_in[5];
  const float* lr_bias    = (const float*)d_in[6];
  const float* senet_w1   = (const float*)d_in[7];
  const float* senet_w2   = (const float*)d_in[8];
  const float* bilinear_W = (const float*)d_in[9];
  const float* W0 = (const float*)d_in[10]; const float* b0 = (const float*)d_in[11];
  const float* W1 = (const float*)d_in[12]; const float* b1 = (const float*)d_in[13];
  const float* W2 = (const float*)d_in[14]; const float* b2 = (const float*)d_in[15];
  const float* W3 = (const float*)d_in[16]; const float* b3 = (const float*)d_in[17];
  float* out = (float*)d_out;

  char* ws = (char*)d_ws; size_t off = 0;
  auto take = [&](size_t bytes) -> void* {
    void* p = ws + off; off = (off + bytes + 255) & ~(size_t)255; return p;
  };
  __bf16* emb_bf  = (__bf16*)take((size_t)B_ * F_ * E_ * 2);
  __bf16* vemb_bf = (__bf16*)take((size_t)B_ * F_ * E_ * 2);
  float*  lr_lg   = (float*) take((size_t)B_ * 4);
  __bf16* Wbt     = (__bf16*)take((size_t)P_ * E_ * E_ * 2);
  __bf16* W0t     = (__bf16*)take((size_t)KC_ * N0_ * 2);
  __bf16* W1t     = (__bf16*)take((size_t)N0_ * N1_ * 2);
  __bf16* W2t     = (__bf16*)take((size_t)N1_ * N2_ * 2);
  __bf16* cbuf    = (__bf16*)take((size_t)B_ * KC_ * 2);
  __bf16* h1      = (__bf16*)take((size_t)B_ * N0_ * 2);
  __bf16* h2      = (__bf16*)take((size_t)B_ * N1_ * 2);
  __bf16* h3      = (__bf16*)take((size_t)B_ * N2_ * 2);
  if (off > ws_size) return;   // insufficient scratch: nothing safe to do

  auto gt = [](long long total) -> unsigned {
    long long g = (total + 255) / 256; return (unsigned)(g > 8192 ? 8192 : g);
  };

  // weight precision conversion + transpose to [N][K] bf16
  k_convert_transpose<<<gt((long long)KC_ * N0_), 256, 0, stream>>>(W0, W0t, KC_, N0_, (long long)KC_ * N0_);
  k_convert_transpose<<<gt((long long)N0_ * N1_), 256, 0, stream>>>(W1, W1t, N0_, N1_, (long long)N0_ * N1_);
  k_convert_transpose<<<gt((long long)N1_ * N2_), 256, 0, stream>>>(W2, W2t, N1_, N2_, (long long)N1_ * N2_);
  k_convert_transpose<<<gt((long long)P_ * E_ * E_), 256, 0, stream>>>(bilinear_W, Wbt, E_, E_, (long long)P_ * E_ * E_);

  k_prep<<<B_, 128, 0, stream>>>(sparse_ids, dense, emb_table, lr_table,
                                 lr_dense_w, lr_bias, senet_w1, senet_w2,
                                 emb_bf, vemb_bf, lr_lg);

  dim3 bb(32, 4), bg(B_ / 16, (PP_ + 3) / 4);
  k_bilinear<<<bg, bb, 0, stream>>>(emb_bf, vemb_bf, Wbt, cbuf);

  k_gemm<<<dim3(N0_ / 64, B_ / 128), 256, 0, stream>>>(
      cbuf, W0t, b0, h1, B_, N0_, KC_, dense, W0 + (size_t)KC_ * N0_, 1);
  k_gemm<<<dim3(N1_ / 64, B_ / 128), 256, 0, stream>>>(
      h1, W1t, b1, h2, B_, N1_, N0_, nullptr, nullptr, 0);
  k_gemm<<<dim3(N2_ / 64, B_ / 128), 256, 0, stream>>>(
      h2, W2t, b2, h3, B_, N2_, N1_, nullptr, nullptr, 0);

  k_zero_loss<<<1, 32, 0, stream>>>(out);
  k_head<<<B_ / 8, 256, 0, stream>>>(h3, W3, b3, lr_lg, label, out);
}
